// MultiHeadAttention_83202106458185
// MI455X (gfx1250) — matrix operand order repaired and checked by reading, not compiled, not run on hardware
//
#include <hip/hip_runtime.h>
#include <hip/hip_bf16.h>

// MI455X / gfx1250, wave32. All matmuls via v_wmma_f32_16x16x32_bf16.
// Pipeline: 3x projection GEMM (fp32 in -> bf16 head-layout out; Q pre-scaled
// by 1/sqrt(HD)), fused masked-score attention (reference has NO softmax:
// masked entries contribute -1e9 * V[k]), final GEMM to fp32 output.
// K-chunk staging in attention uses gfx1250 global_load_async_to_lds_b128.

#define DEV __device__ __forceinline__

typedef __attribute__((ext_vector_type(16))) __bf16          v16bf;
typedef __attribute__((ext_vector_type(8)))  __bf16          v8bf;
typedef __attribute__((ext_vector_type(8)))  float           v8f;
typedef __attribute__((ext_vector_type(8)))  float           float8;
typedef __attribute__((ext_vector_type(8)))  unsigned short  u16x8;

struct BFPack { u16x8 lo, hi; };

DEV v16bf make_frag(u16x8 lo, u16x8 hi) {
  BFPack p{lo, hi};
  return __builtin_bit_cast(v16bf, p);
}

// packed fp32x8 -> bf16x8 (native v_cvt path) as raw halves
DEV u16x8 cvt8(float8 f) {
  return __builtin_bit_cast(u16x8, __builtin_convertvector(f, v8bf));
}
DEV unsigned short f2bf(float f) {
  return __builtin_bit_cast(unsigned short, (__bf16)f);
}

constexpr int BATCH = 4, S = 2048, D = 1024, H = 16, HD = 64;
constexpr int M_TOT = BATCH * S;          // 8192 token rows
constexpr int BM = 128, BN = 128, BK = 32;
constexpr int LDA = BK + 8;               // 40 halves = 80B row stride (16B aligned, bank-spread)

// Convert 16 fp32 -> 16 bf16, store as two 16B LDS vectors (-> ds_store_b128)
DEV void stage16_f32(const float* __restrict__ src, unsigned short* __restrict__ dst) {
  float8 f0 = ((const float8*)src)[0];
  float8 f1 = ((const float8*)src)[1];
  ((u16x8*)dst)[0] = cvt8(f0);
  ((u16x8*)dst)[1] = cvt8(f1);
}

// MODE 0: A = fp32 [M,D], out = bf16 head layout [B,H,S,HD], scaled  (Q/K/V proj)
// MODE 2: A = bf16 [M,D], out = fp32 flat [M,D]                      (output proj)
// Computes out[m,n] = (sum_k A[m,k]*W[n,k] + bias[n]) * out_scale    (X @ W^T + b)
template <int MODE>
__global__ __launch_bounds__(256) void gemm_wmma(
    const void* __restrict__ Av, const float* __restrict__ W,
    const float* __restrict__ bias, void* __restrict__ Outv, float out_scale) {
  __shared__ unsigned short As[BM * LDA];
  __shared__ unsigned short Bs[BN * LDA];

  const int tid  = threadIdx.x;
  const int lane = tid & 31;
  const int wave = tid >> 5;          // 8 waves (wave32)
  const int wm   = wave >> 1;         // 4x2 wave grid: each wave 32x64
  const int wn   = wave & 1;
  const int m0   = blockIdx.y * BM;
  const int n0   = blockIdx.x * BN;

  v8f acc[2][4];
#pragma unroll
  for (int i = 0; i < 2; i++)
#pragma unroll
    for (int j = 0; j < 4; j++) acc[i][j] = {};

  const int arow = tid >> 1;          // 128 rows, 2 threads/row
  const int acs  = (tid & 1) * 16;    // 16 elements each

  for (int kt = 0; kt < D; kt += BK) {
    // ---- stage A tile (128x32) ----
    if constexpr (MODE == 0) {
      stage16_f32((const float*)Av + (size_t)(m0 + arow) * D + kt + acs,
                  &As[arow * LDA + acs]);
    } else {
      const unsigned short* src =
          (const unsigned short*)Av + (size_t)(m0 + arow) * D + kt + acs;
      ((u16x8*)&As[arow * LDA + acs])[0] = ((const u16x8*)src)[0];
      ((u16x8*)&As[arow * LDA + acs])[1] = ((const u16x8*)src)[1];
    }
    // ---- stage B tile: W rows n (W stored [N][K], exactly W^T columns) ----
    stage16_f32(W + (size_t)(n0 + arow) * D + kt + acs, &Bs[arow * LDA + acs]);

    if (kt + BK < D) {  // gfx1250 global_prefetch_b8 for next K tile
      __builtin_prefetch(W + (size_t)(n0 + arow) * D + kt + BK + acs, 0, 0);
      if constexpr (MODE == 0)
        __builtin_prefetch((const float*)Av + (size_t)(m0 + arow) * D + kt + BK + acs, 0, 0);
      else
        __builtin_prefetch((const unsigned short*)Av + (size_t)(m0 + arow) * D + kt + BK + acs, 0, 0);
    }
    __syncthreads();

    // ---- fragments: contiguous k order (k = 16*half + s), same for A and B ----
    const int kb = (lane < 16) ? 0 : 16;   // A: K base per half-wave
    v16bf af[2];
#pragma unroll
    for (int i = 0; i < 2; i++) {
      int r = wm * 32 + i * 16 + (lane & 15);
      af[i] = make_frag(*(const u16x8*)&As[r * LDA + kb],
                        *(const u16x8*)&As[r * LDA + kb + 8]);
    }
    const int ks = (lane < 16) ? 0 : 16;   // B: K base per half-wave
    v16bf bf[4];
#pragma unroll
    for (int j = 0; j < 4; j++) {
      int n = wn * 64 + j * 16 + (lane & 15);
      bf[j] = make_frag(*(const u16x8*)&Bs[n * LDA + ks],
                        *(const u16x8*)&Bs[n * LDA + ks + 8]);
    }
#pragma unroll
    for (int i = 0; i < 2; i++)
#pragma unroll
      for (int j = 0; j < 4; j++)
        acc[i][j] = __builtin_amdgcn_wmma_f32_16x16x32_bf16(
            false, af[i], false, bf[j], (short)0, acc[i][j], false, false);
    __syncthreads();
  }

  // ---- epilogue: C layout = lane(l&15)=col, VGPR r -> row r + (l>=16?8:0) ----
  const int rhi = (lane >= 16) ? 8 : 0;
#pragma unroll
  for (int i = 0; i < 2; i++) {
#pragma unroll
    for (int j = 0; j < 4; j++) {
      int gn = n0 + wn * 64 + j * 16 + (lane & 15);
      float bv = bias[gn];
#pragma unroll
      for (int r = 0; r < 8; r++) {
        int gm = m0 + wm * 32 + i * 16 + r + rhi;
        float val = (acc[i][j][r] + bv) * out_scale;
        if constexpr (MODE == 0) {
          int bb = gm >> 11, s = gm & (S - 1);   // gm = bb*S + s
          int h = gn >> 6, hd = gn & 63;         // gn = h*HD + hd
          ((unsigned short*)Outv)[((size_t)(bb * H + h) * S + s) * HD + hd] = f2bf(val);
        } else {
          ((float*)Outv)[(size_t)gm * D + gn] = val;
        }
      }
    }
  }
}

// Attention without softmax: out[q] = sum_k score(q,k) * V[k] where
// score = (Q.K^T)/8 below/on diagonal and -1e9 above it (full S x S matmul;
// Q is pre-scaled by 0.125 in its projection).
__global__ __launch_bounds__(256) void attn_wmma(
    const unsigned short* __restrict__ Qh, const unsigned short* __restrict__ Kh,
    const unsigned short* __restrict__ Vh, unsigned short* __restrict__ Oh) {
  constexpr int LDK = 72;  // 144B row stride, 16B aligned
  constexpr int LDV = 40;  // 80B
  constexpr int LDS_SC = 40;
  __shared__ unsigned short Ks[32 * LDK];        // K chunk: [key][hd]
  __shared__ unsigned short Vs[HD * LDV];        // V chunk transposed: [hd][key]
  __shared__ unsigned short Sc[8 * 16 * LDS_SC]; // per-wave score re-layout scratch

  const int tid = threadIdx.x, lane = tid & 31, wave = tid >> 5;
  const int bh = blockIdx.y;                 // b*H + h
  const int q0 = blockIdx.x * 128;           // 128 queries per block, 16 per wave
  const size_t base = (size_t)bh * S * HD;
  const int qrow_lo = q0 + wave * 16;
  const int rhi = (lane >= 16) ? 8 : 0;
  const int kb = (lane < 16) ? 0 : 16;       // A-frag K base
  const int ks16 = (lane < 16) ? 0 : 16;     // B-frag K base

  // Q fragments (A layout, 16 q-rows x 64 hd) held in registers for whole pass
  v16bf qf[2];
  {
    const int qr = qrow_lo + (lane & 15);
#pragma unroll
    for (int sseg = 0; sseg < 2; sseg++) {
      const unsigned short* p = Qh + base + (size_t)qr * HD + sseg * 32 + kb;
      qf[sseg] = make_frag(*(const u16x8*)p, *(const u16x8*)(p + 8));
    }
  }

  v8f oacc[4];
#pragma unroll
  for (int j = 0; j < 4; j++) oacc[j] = {};

  unsigned short* mySc = &Sc[wave * 16 * LDS_SC];
  const int skey = tid >> 3, sseg8 = (tid & 7) * 8;  // staging split: 16B/thread

  for (int k0 = 0; k0 < S; k0 += 32) {
    // ---- stage K chunk [32][64] via gfx1250 async LDS load (ASYNCcnt) ----
    {
      unsigned lds_addr = (unsigned)(uintptr_t)&Ks[skey * LDK + sseg8];
      unsigned long long gaddr =
          (unsigned long long)(uintptr_t)(Kh + base + (size_t)(k0 + skey) * HD + sseg8);
      asm volatile("global_load_async_to_lds_b128 %0, %1, off"
                   :: "v"(lds_addr), "v"(gaddr) : "memory");
    }
    // ---- stage V chunk transposed [hd][key] through VGPRs ----
    {
      u16x8 v = *(const u16x8*)(Vh + base + (size_t)(k0 + skey) * HD + sseg8);
#pragma unroll
      for (int e = 0; e < 8; e++) Vs[(sseg8 + e) * LDV + skey] = v[e];
    }
    asm volatile("s_wait_asynccnt 0" ::: "memory");
    __syncthreads();

    // preload V B-fragments (B layout: lane=hd col, 16 contiguous keys)
    v16bf vf[4];
#pragma unroll
    for (int j = 0; j < 4; j++) {
      const unsigned short* vp = &Vs[(j * 16 + (lane & 15)) * LDV + ks16];
      vf[j] = make_frag(*(const u16x8*)vp, *(const u16x8*)(vp + 8));
    }

    // scores: two 16-key tiles, K-dim = HD = 64 -> 2 WMMAs each
#pragma unroll
    for (int half = 0; half < 2; half++) {
      v8f sacc = {};
#pragma unroll
      for (int sseg = 0; sseg < 2; sseg++) {
        const unsigned short* p =
            &Ks[(half * 16 + (lane & 15)) * LDK + sseg * 32 + ks16];
        v16bf kf = make_frag(*(const u16x8*)p, *(const u16x8*)(p + 8));
        sacc = __builtin_amdgcn_wmma_f32_16x16x32_bf16(
            false, qf[sseg], false, kf, (short)0, sacc, false, false);
      }
      // causal mask: masked entries ARE -1e9 (reference has no softmax)
#pragma unroll
      for (int r = 0; r < 8; r++) {
        int row = r + rhi;
        int qg = qrow_lo + row;
        int kg = k0 + half * 16 + (lane & 15);
        float v = (kg <= qg) ? sacc[r] : -1e9f;
        mySc[row * LDS_SC + half * 16 + (lane & 15)] = f2bf(v);
      }
    }
    __syncthreads();  // scores visible; also fences Ks/Vs before next staging

    // score A-fragment (16 q x 32 keys) and 4 output WMMAs (hd segments)
    const unsigned short* sp = &mySc[(lane & 15) * LDS_SC + kb];
    v16bf sf = make_frag(*(const u16x8*)sp, *(const u16x8*)(sp + 8));
#pragma unroll
    for (int j = 0; j < 4; j++)
      oacc[j] = __builtin_amdgcn_wmma_f32_16x16x32_bf16(
          false, sf, false, vf[j], (short)0, oacc[j], false, false);
  }

  // write attention output as bf16 [B, S, D] (row s, col h*64+hd)
  const int b = bh >> 4, h = bh & 15;
#pragma unroll
  for (int j = 0; j < 4; j++) {
#pragma unroll
    for (int r = 0; r < 8; r++) {
      int qg = qrow_lo + r + rhi;
      int hd = j * 16 + (lane & 15);
      Oh[((size_t)(b * S + qg)) * D + h * HD + hd] = f2bf(oacc[j][r]);
    }
  }
}

extern "C" void kernel_launch(void* const* d_in, const int* in_sizes, int n_in,
                              void* d_out, int out_size, void* d_ws, size_t ws_size,
                              hipStream_t stream) {
  const float* q_in = (const float*)d_in[0];
  const float* k_in = (const float*)d_in[1];
  const float* v_in = (const float*)d_in[2];
  // d_in[3] = causal mask (int32) — computed analytically in-kernel
  const float* Wq = (const float*)d_in[4];
  const float* bq = (const float*)d_in[5];
  const float* Wk = (const float*)d_in[6];
  const float* bk = (const float*)d_in[7];
  const float* Wv = (const float*)d_in[8];
  const float* bv = (const float*)d_in[9];
  const float* Wo = (const float*)d_in[10];
  const float* bo = (const float*)d_in[11];

  // workspace: 4 bf16 buffers of M_TOT*D halves = 16MB each (64MB total)
  unsigned short* Qh = (unsigned short*)d_ws;
  unsigned short* Kh = Qh + (size_t)M_TOT * D;
  unsigned short* Vh = Kh + (size_t)M_TOT * D;
  unsigned short* AO = Vh + (size_t)M_TOT * D;

  dim3 blk(256);
  dim3 ggrid(D / BN, M_TOT / BM);  // 8 x 64

  gemm_wmma<0><<<ggrid, blk, 0, stream>>>(q_in, Wq, bq, Qh, 0.125f);  // 1/sqrt(HD) folded
  gemm_wmma<0><<<ggrid, blk, 0, stream>>>(k_in, Wk, bk, Kh, 1.0f);
  gemm_wmma<0><<<ggrid, blk, 0, stream>>>(v_in, Wv, bv, Vh, 1.0f);
  attn_wmma<<<dim3(S / 128, BATCH * H), blk, 0, stream>>>(Qh, Kh, Vh, AO);
  gemm_wmma<2><<<ggrid, blk, 0, stream>>>(AO, Wo, bo, (float*)d_out, 1.0f);
}
